// WaveNetRNNClassifier_17446157156591
// MI455X (gfx1250) — matrix operand order repaired and checked by reading, not compiled, not run on hardware
//
#include <hip/hip_runtime.h>
#include <hip/hip_bf16.h>
#include <math.h>

// ---------------------------------------------------------------------------
// WaveNet + 2-layer LSTM + attention classifier for MI455X (gfx1250, wave32)
// All dense matmuls on v_wmma_f32_16x16x32_f16. Every hot-loop WMMA operand
// is a single 32B vector load from a pre-packed (lane-layout) f16 blob.
// ---------------------------------------------------------------------------

typedef __attribute__((ext_vector_type(16))) _Float16 v16h;
typedef __attribute__((ext_vector_type(8)))  float    v8f;

#define DEV __device__ __forceinline__

static constexpr int Bn = 8;
static constexpr int Tn = 4096;
static constexpr int Hn = 64;
static constexpr int Pn = Bn * Tn;        // 32768 positions

DEV float sigf(float x) { return 1.0f / (1.0f + expf(-x)); }

DEV v8f WMMA(v16h a, v16h b, v8f c) {
  return __builtin_amdgcn_wmma_f32_16x16x32_f16(false, a, false, b,
                                                (short)0, c, false, false);
}

// ---- WMMA lane layouts -----------------------------------------------------
// A (16x32 f16): lane holds row m=lane&15; element i -> K = a_k(i,lane)
DEV int a_k(int i, int lane) {
  return ((lane & 16) ? 16 : 0) + i;
}
// B (32x16 f16): lane holds col n=lane&15; element i -> K = (lane&16?16:0)+i
// C/D (16x16 f32): col n=lane&15, row = r + (lane&16 ? 8 : 0)

// writer-side flat index into an A-packed blob [Mtiles][Kchunks][32][16]
DEV int apk_idx(int Kchunks, int m, int k) {
  int kc = k >> 5, rem = k & 31;
  int lane = (m & 15) + ((rem & 16) ? 16 : 0);
  int i = rem & 15;
  return (((m >> 4) * Kchunks + kc) * 32 + lane) * 16 + i;
}
// writer-side flat index into a B-packed blob [Kchunks][Ntiles][32][16]
DEV int bpk_idx(int Ntiles, int k, int nt, int nn) {
  int kc = k >> 5, rem = k & 31;
  int lane = nn + ((rem & 16) ? 16 : 0);
  int i = rem & 15;
  return ((kc * Ntiles + nt) * 32 + lane) * 16 + i;
}

// ---- global activation blobs in lane layout -------------------------------
// B-packed by column tile: blob[[c>>4][kc][ (c&15)+khalf ][16]]
DEV size_t bcol_idx(int chan, int c) {
  int kc = chan >> 5, rem = chan & 31;
  return ((size_t)((c >> 4) * 2 + kc) * 32 + (c & 15) + (rem & 16)) * 16 +
         (rem & 15);
}
// A-packed by row tile: blob[[m>>4][kc][lane][16]]
DEV size_t arow_idx(int m, int chan) {
  int kc = chan >> 5, rem = chan & 31;
  int lane = (m & 15) + ((rem & 16) ? 16 : 0);
  int i = rem & 15;
  return ((size_t)((m >> 4) * 2 + kc) * 32 + lane) * 16 + i;
}

// B operand from a column-packed blob; c0 may be unaligned/negative (dilated
// taps) -- the per-lane tile/slot index handles any shift with 1 vector load.
DEV v16h ldB_pkcol(const _Float16* blob, int kc, int c0, int lane) {
  int c = c0 + (lane & 15);
  if (c < 0) {
    v16h z;
#pragma unroll
    for (int i = 0; i < 16; ++i) z[i] = (_Float16)0.0f;
    return z;
  }
  return ((const v16h*)blob)[((c >> 4) * 2 + kc) * 32 + (c & 15) + (lane & 16)];
}

// A operand from a row-packed blob; m0 must be 16-aligned.
DEV v16h ldA_pkrow(const _Float16* blob, int kc, int m0, int lane) {
  return ((const v16h*)blob)[((m0 >> 4) * 2 + kc) * 32 + lane];
}

// ---- f32 fallbacks (run-once kernels only) --------------------------------
DEV v16h ldB_f32(const float* base, long sK, long sN, int k0, int n0, int lane) {
  int n = n0 + (lane & 15);
  int kb = k0 + ((lane & 16) ? 16 : 0);
  v16h r;
#pragma unroll
  for (int i = 0; i < 16; ++i)
    r[i] = (_Float16)base[(long)(kb + i) * sK + (long)n * sN];
  return r;
}

DEV v16h ldB_relu_f32(const float* base, long sK, long sN, int k0, int n0, int lane) {
  int n = n0 + (lane & 15);
  int kb = k0 + ((lane & 16) ? 16 : 0);
  v16h r;
#pragma unroll
  for (int i = 0; i < 16; ++i)
    r[i] = (_Float16)fmaxf(base[(long)(kb + i) * sK + (long)n * sN], 0.0f);
  return r;
}

DEV v8f ldC_bias_row(const float* bias, int m0, int lane) {
  int mo = (lane & 16) ? 8 : 0;
  v8f c;
#pragma unroll
  for (int r = 0; r < 8; ++r) c[r] = bias[m0 + mo + r];
  return c;
}

DEV v8f ldC_bias_col(const float* bias, int n0, int lane) {
  float v = bias[n0 + (lane & 15)];
  v8f c;
#pragma unroll
  for (int r = 0; r < 8; ++r) c[r] = v;
  return c;
}

// ===========================================================================
// Weight pre-swizzle kernels (run once; blobs stay resident in L2)
// ===========================================================================

__global__ void k_packA(const float* __restrict__ src, long sM, long sK,
                        int Mtiles, int Kchunks, _Float16* __restrict__ dst) {
  int idx = blockIdx.x * 256 + threadIdx.x;
  int total = Mtiles * Kchunks * 32;
  if (idx >= total) return;
  int lane = idx & 31, tc = idx >> 5;
  int kc = tc % Kchunks, mt = tc / Kchunks;
  int m = mt * 16 + (lane & 15);
  _Float16* d = dst + (size_t)idx * 16;
#pragma unroll
  for (int i = 0; i < 16; ++i)
    d[i] = (_Float16)src[(long)m * sM + (long)(kc * 32 + a_k(i, lane)) * sK];
}

__global__ void k_packB(const float* __restrict__ src, long sK, long sN,
                        int Kchunks, int Ntiles, _Float16* __restrict__ dst) {
  int idx = blockIdx.x * 256 + threadIdx.x;
  int total = Kchunks * Ntiles * 32;
  if (idx >= total) return;
  int lane = idx & 31, tc = idx >> 5;
  int nt = tc % Ntiles, kc = tc / Ntiles;
  int n = nt * 16 + (lane & 15);
  int kb = kc * 32 + ((lane & 16) ? 16 : 0);
  _Float16* d = dst + (size_t)idx * 16;
#pragma unroll
  for (int i = 0; i < 16; ++i)
    d[i] = (_Float16)src[(long)(kb + i) * sK + (long)n * sN];
}

// ===========================================================================
// Pipeline kernels
// ===========================================================================

__global__ void k_zero(float* p, int n) {
  int i = blockIdx.x * blockDim.x + threadIdx.x;
  if (i < n) p[i] = 0.0f;
}

// x[B,T,32] -> resid f32 [B,64,T] + f16 mirror (B-packed by column tile)
__global__ void __launch_bounds__(256) k_inproj(const float* __restrict__ x,
                                                const _Float16* __restrict__ pkW,
                                                const float* __restrict__ bias,
                                                float* __restrict__ resid,
                                                _Float16* __restrict__ resid_h) {
  const int TCH = Tn / 64;
  int b = blockIdx.x / TCH, t0 = (blockIdx.x % TCH) * 64;
  int lane = threadIdx.x & 31, wave = threadIdx.x >> 5;
  const float* xb = x + ((size_t)b * Tn + t0) * 32;
  float* rb = resid + (size_t)b * Hn * Tn;
  _Float16* rh = resid_h + (size_t)b * Hn * Tn;
  const v16h* wv = (const v16h*)pkW;          // [4 mt][1 kc][32]
#pragma unroll
  for (int s = 0; s < 2; ++s) {
    int tile = wave * 2 + s;
    int m0 = (tile >> 2) * 16, n0 = (tile & 3) * 16;
    v8f acc = ldC_bias_row(bias, m0, lane);
    acc = WMMA(wv[(m0 >> 4) * 32 + lane], ldB_f32(xb, 1, 32, 0, n0, lane), acc);
    int nn = lane & 15, mo = (lane & 16) ? 8 : 0;
#pragma unroll
    for (int r = 0; r < 8; ++r) {
      int m = m0 + mo + r, t = t0 + n0 + nn;
      rb[(size_t)m * Tn + t] = acc[r];
      rh[bcol_idx(m, t)] = (_Float16)acc[r];
    }
  }
}

// One WaveNet residual block. wpk = 6 packed A matrices:
// [filt tap1, filt tap0, gate tap1, gate tap0, res_w, skip_w], each 4096 f16.
__global__ void __launch_bounds__(256) k_block(
    const float* __restrict__ rin, const _Float16* __restrict__ rin_h,
    float* __restrict__ rout, _Float16* __restrict__ rout_h,
    float* __restrict__ skip, const _Float16* __restrict__ wpk,
    const float* __restrict__ fb, const float* __restrict__ gb,
    const float* __restrict__ rbias, const float* __restrict__ sbias,
    int dil) {
  __shared__ v16h gatedB[2 * 4 * 32];       // B-packed gated [kc][nt][lane]
  _Float16* g16 = (_Float16*)gatedB;
  const int TCH = Tn / 64;
  int b = blockIdx.x / TCH, t0 = (blockIdx.x % TCH) * 64;
  int lane = threadIdx.x & 31, wave = threadIdx.x >> 5;
  const _Float16* rh = rin_h + (size_t)b * Hn * Tn;
  const v16h* wv = (const v16h*)wpk;        // matrix j at j*256, tile (mt*2+kc)*32

  int tpre = t0 - dil;                       // warm cache for dilated taps
  if (tpre >= 0)
    __builtin_prefetch(rh + (size_t)((tpre >> 4) * 2) * 512 + lane * 32, 0, 1);

#pragma unroll
  for (int s = 0; s < 2; ++s) {
    int tile = wave * 2 + s;
    int mt = tile >> 2, nt = tile & 3;
    int m0 = mt * 16, n0 = nt * 16;
    v8f accf = ldC_bias_row(fb, m0, lane);
    v8f accg = ldC_bias_row(gb, m0, lane);
#pragma unroll
    for (int kc = 0; kc < 2; ++kc) {
      v16h bcur = ldB_pkcol(rh, kc, t0 + n0, lane);
      v16h bsh  = ldB_pkcol(rh, kc, t0 + n0 - dil, lane);
      int ti = (mt * 2 + kc) * 32 + lane;
      accf = WMMA(wv[0 * 256 + ti], bcur, accf);
      accf = WMMA(wv[1 * 256 + ti], bsh,  accf);
      accg = WMMA(wv[2 * 256 + ti], bcur, accg);
      accg = WMMA(wv[3 * 256 + ti], bsh,  accg);
    }
    int nn = lane & 15, mo = (lane & 16) ? 8 : 0;
#pragma unroll
    for (int r = 0; r < 8; ++r) {
      float gv = tanhf(accf[r]) * sigf(accg[r]);
      g16[bpk_idx(4, m0 + mo + r, nt, nn)] = (_Float16)gv;   // (k=chan, n=pos)
    }
  }
  __syncthreads();

  const float* rf = rin + (size_t)b * Hn * Tn;
  float* ro = rout + (size_t)b * Hn * Tn;
  _Float16* roh = rout_h + (size_t)b * Hn * Tn;
  float* sk = skip + (size_t)b * Hn * Tn;
#pragma unroll
  for (int s = 0; s < 2; ++s) {
    int tile = wave * 2 + s;
    int mt = tile >> 2, nt = tile & 3;
    int m0 = mt * 16, n0 = nt * 16;
    v8f accr = ldC_bias_row(rbias, m0, lane);
    v8f accs = ldC_bias_row(sbias, m0, lane);
#pragma unroll
    for (int kc = 0; kc < 2; ++kc) {
      v16h bg = gatedB[(kc * 4 + nt) * 32 + lane];
      int ti = (mt * 2 + kc) * 32 + lane;
      accr = WMMA(wv[4 * 256 + ti], bg, accr);
      accs = WMMA(wv[5 * 256 + ti], bg, accs);
    }
    int nn = lane & 15, mo = (lane & 16) ? 8 : 0;
#pragma unroll
    for (int r = 0; r < 8; ++r) {
      int m = m0 + mo + r, t = t0 + n0 + nn;
      float rv = rf[(size_t)m * Tn + t] + accr[r];
      ro[(size_t)m * Tn + t] = rv;
      roh[bcol_idx(m, t)] = (_Float16)rv;
      sk[(size_t)m * Tn + t] += accs[r];
    }
  }
}

// skip[B,64,T] -> relu -> w1 -> relu -> w2 -> hwav_h (A-packed by position)
__global__ void __launch_bounds__(256) k_post(const float* __restrict__ skip,
                                              const _Float16* __restrict__ pkW1,
                                              const float* __restrict__ b1,
                                              const _Float16* __restrict__ pkW2,
                                              const float* __restrict__ b2,
                                              _Float16* __restrict__ hwav_h) {
  __shared__ v16h s1B[2 * 4 * 32];          // B-packed hidden [kc][nt][lane]
  _Float16* s16 = (_Float16*)s1B;
  const int TCH = Tn / 64;
  int b = blockIdx.x / TCH, t0 = (blockIdx.x % TCH) * 64;
  int lane = threadIdx.x & 31, wave = threadIdx.x >> 5;
  const float* skip_b = skip + (size_t)b * Hn * Tn;
  const v16h* w1v = (const v16h*)pkW1;
  const v16h* w2v = (const v16h*)pkW2;
#pragma unroll
  for (int s = 0; s < 2; ++s) {
    int tile = wave * 2 + s;
    int mt = tile >> 2, nt = tile & 3;
    int m0 = mt * 16, n0 = nt * 16;
    v8f acc = ldC_bias_row(b1, m0, lane);
#pragma unroll
    for (int kc = 0; kc < 2; ++kc)
      acc = WMMA(w1v[(mt * 2 + kc) * 32 + lane],
                 ldB_relu_f32(skip_b + t0, Tn, 1, kc * 32, n0, lane), acc);
    int nn = lane & 15, mo = (lane & 16) ? 8 : 0;
#pragma unroll
    for (int r = 0; r < 8; ++r)
      s16[bpk_idx(4, m0 + mo + r, nt, nn)] = (_Float16)fmaxf(acc[r], 0.0f);
  }
  __syncthreads();
#pragma unroll
  for (int s = 0; s < 2; ++s) {
    int tile = wave * 2 + s;
    int mt = tile >> 2, nt = tile & 3;
    int m0 = mt * 16, n0 = nt * 16;
    v8f acc = ldC_bias_row(b2, m0, lane);
#pragma unroll
    for (int kc = 0; kc < 2; ++kc)
      acc = WMMA(w2v[(mt * 2 + kc) * 32 + lane],
                 s1B[(kc * 4 + nt) * 32 + lane], acc);
    int nn = lane & 15, mo = (lane & 16) ? 8 : 0;
#pragma unroll
    for (int r = 0; r < 8; ++r) {
      int pos = b * Tn + t0 + n0 + nn;          // row = position
      hwav_h[arow_idx(pos, m0 + mo + r)] = (_Float16)acc[r];
    }
  }
}

// xg[t][batch][256] = act_h[pos,64] @ Wih^T + bih + bhh
__global__ void __launch_bounds__(256) k_xg(const _Float16* __restrict__ act_h,
                                            const _Float16* __restrict__ pkWih,
                                            const float* __restrict__ bih,
                                            const float* __restrict__ bhh,
                                            float* __restrict__ xg) {
  int pos0 = blockIdx.x * 32;
  int lane = threadIdx.x & 31, wave = threadIdx.x >> 5;
  const v16h* wv = (const v16h*)pkWih;      // [2 kc][16 nt][32]
#pragma unroll
  for (int s = 0; s < 4; ++s) {
    int tile = wave * 4 + s;                 // 32 tiles: 2 (pos) x 16 (gates)
    int m0 = (tile >> 4) * 16, nt = tile & 15, n0 = nt * 16;
    int nn = lane & 15;
    float bv = bih[n0 + nn] + bhh[n0 + nn];
    v8f acc;
#pragma unroll
    for (int r = 0; r < 8; ++r) acc[r] = bv;
#pragma unroll
    for (int kc = 0; kc < 2; ++kc)
      acc = WMMA(ldA_pkrow(act_h, kc, pos0 + m0, lane),
                 wv[(kc * 16 + nt) * 32 + lane], acc);
    int mo = (lane & 16) ? 8 : 0;
#pragma unroll
    for (int r = 0; r < 8; ++r) {
      int pos = pos0 + m0 + mo + r;
      int t = pos & (Tn - 1), bb = pos >> 12;  // Tn == 4096
      xg[((size_t)t * 8 + bb) * 256 + n0 + nn] = acc[r];
    }
  }
}

// Sequential LSTM recurrence: one persistent workgroup, 16 waves, h in
// A-packed f16 LDS so every step is exactly 2 LDS vector loads + 2 WMMAs/wave.
__global__ void __launch_bounds__(512) k_lstm_rec(const float* __restrict__ xg,
                                                  const _Float16* __restrict__ pkWhh,
                                                  _Float16* __restrict__ out_h) {
  __shared__ v16h hAv[2 * 32];              // A-packed h [kc][lane]
  __shared__ float cbuf[8 * 64];
  __shared__ float gbuf[16 * 256];
  _Float16* hA = (_Float16*)hAv;
  int tid = threadIdx.x, lane = tid & 31, wave = tid >> 5;
  for (int i = tid; i < 2 * 32 * 16; i += 512) hA[i] = (_Float16)0.0f;
  for (int i = tid; i < 8 * 64; i += 512) cbuf[i] = 0.0f;
  __syncthreads();

  int g0 = wave * 16;
  const v16h* wv = (const v16h*)pkWhh;      // [2 kc][16 nt][32]
  v16h whh0 = wv[(0 * 16 + wave) * 32 + lane];
  v16h whh1 = wv[(1 * 16 + wave) * 32 + lane];

  int nn = lane & 15, mo = (lane & 16) ? 8 : 0;
  for (int t = 0; t < Tn; ++t) {
    const float* xgt = xg + (size_t)t * 8 * 256;   // all batches contiguous
    v8f acc;
#pragma unroll
    for (int r = 0; r < 8; ++r) {
      int bb = r + mo;
      acc[r] = (bb < 8) ? xgt[bb * 256 + g0 + nn] : 0.0f;
    }
    acc = WMMA(hAv[0 * 32 + lane], whh0, acc);
    acc = WMMA(hAv[1 * 32 + lane], whh1, acc);
#pragma unroll
    for (int r = 0; r < 8; ++r) gbuf[(r + mo) * 256 + g0 + nn] = acc[r];
    __syncthreads();
    {
      int bb = tid >> 6, j = tid & 63;       // 512 = 8 batch x 64 hidden
      float gi = gbuf[bb * 256 + j];
      float gf = gbuf[bb * 256 + 64 + j];
      float gg = gbuf[bb * 256 + 128 + j];
      float go = gbuf[bb * 256 + 192 + j];
      float c = sigf(gf) * cbuf[bb * 64 + j] + sigf(gi) * tanhf(gg);
      float h = sigf(go) * tanhf(c);
      cbuf[bb * 64 + j] = c;
      hA[apk_idx(2, bb, j)] = (_Float16)h;   // back into A-packed layout
      out_h[arow_idx(bb * Tn + t, j)] = (_Float16)h;
    }
    __syncthreads();
  }
}

// fused = relu([hwav|l1|l2] @ fus_w1^T + b1) @ fus_w2^T + b2
// outputs: fused f32 [pos][64] (for pooling) + fused_h (B-packed by position)
__global__ void __launch_bounds__(256) k_fusion(const _Float16* __restrict__ hwav_h,
                                                const _Float16* __restrict__ l1_h,
                                                const _Float16* __restrict__ l2_h,
                                                const _Float16* __restrict__ pkW1,
                                                const float* __restrict__ b1,
                                                const _Float16* __restrict__ pkW2,
                                                const float* __restrict__ b2,
                                                float* __restrict__ fused,
                                                _Float16* __restrict__ fused_h) {
  __shared__ v16h s1A[4 * 2 * 32];          // A-packed hidden [mt][kc][lane]
  _Float16* s16 = (_Float16*)s1A;
  int pos0 = blockIdx.x * 64;
  int lane = threadIdx.x & 31, wave = threadIdx.x >> 5;
  const _Float16* srcs[3] = {hwav_h, l1_h, l2_h};
  const v16h* w1v = (const v16h*)pkW1;      // [6 kc][4 nt][32]
  const v16h* w2v = (const v16h*)pkW2;      // [2 kc][4 nt][32]
#pragma unroll
  for (int s = 0; s < 2; ++s) {
    int tile = wave * 2 + s;
    int mt = tile >> 2, nt = tile & 3;      // M = positions, N = outputs
    int m0 = mt * 16, n0 = nt * 16;
    v8f acc = ldC_bias_col(b1, n0, lane);
    for (int ch = 0; ch < 6; ++ch) {
      acc = WMMA(ldA_pkrow(srcs[ch >> 1], ch & 1, pos0 + m0, lane),
                 w1v[(ch * 4 + nt) * 32 + lane], acc);
    }
    int nn = lane & 15, mo = (lane & 16) ? 8 : 0;
#pragma unroll
    for (int r = 0; r < 8; ++r)
      s16[apk_idx(2, m0 + mo + r, n0 + nn)] = (_Float16)fmaxf(acc[r], 0.0f);
  }
  __syncthreads();
#pragma unroll
  for (int s = 0; s < 2; ++s) {
    int tile = wave * 2 + s;
    int mt = tile >> 2, nt = tile & 3;
    int m0 = mt * 16, n0 = nt * 16;
    v8f acc = ldC_bias_col(b2, n0, lane);
#pragma unroll
    for (int kc = 0; kc < 2; ++kc)
      acc = WMMA(s1A[(mt * 2 + kc) * 32 + lane],
                 w2v[(kc * 4 + nt) * 32 + lane], acc);
    int nn = lane & 15, mo = (lane & 16) ? 8 : 0;
#pragma unroll
    for (int r = 0; r < 8; ++r) {
      int pos = pos0 + m0 + mo + r, chan = n0 + nn;
      fused[(size_t)pos * 64 + chan] = acc[r];
      fused_h[bcol_idx(chan, pos)] = (_Float16)acc[r];   // col = position
    }
  }
}

// att[pos] = att_w2 . tanh(att_w1 @ fused[pos] + b1) + b2
__global__ void __launch_bounds__(256) k_att(const _Float16* __restrict__ fused_h,
                                             const _Float16* __restrict__ pkW1,
                                             const float* __restrict__ b1,
                                             const float* __restrict__ w2,
                                             const float* __restrict__ b2,
                                             float* __restrict__ att) {
  __shared__ float vbuf[64 * 68];           // [chan][pos]
  int pos0 = blockIdx.x * 64;
  int lane = threadIdx.x & 31, wave = threadIdx.x >> 5;
  const v16h* w1v = (const v16h*)pkW1;      // [4 mt][2 kc][32]
#pragma unroll
  for (int s = 0; s < 2; ++s) {
    int tile = wave * 2 + s;
    int mt = tile >> 2, nt = tile & 3;      // M = out chan, N = pos
    int m0 = mt * 16, n0 = nt * 16;
    v8f acc = ldC_bias_row(b1, m0, lane);
#pragma unroll
    for (int kc = 0; kc < 2; ++kc)
      acc = WMMA(w1v[(mt * 2 + kc) * 32 + lane],
                 ldB_pkcol(fused_h, kc, pos0 + n0, lane), acc);
    int nn = lane & 15, mo = (lane & 16) ? 8 : 0;
#pragma unroll
    for (int r = 0; r < 8; ++r)
      vbuf[(m0 + mo + r) * 68 + n0 + nn] = tanhf(acc[r]);
  }
  __syncthreads();
  int tid = threadIdx.x;
  if (tid < 64) {
    float a = b2[0];
    for (int m = 0; m < 64; ++m) a += w2[m] * vbuf[m * 68 + tid];
    att[pos0 + tid] = a;
  }
}

// per-batch masked softmax over T + attention pooling -> pooled[b,64]
__global__ void __launch_bounds__(256) k_pool(const float* __restrict__ fused,
                                              const float* __restrict__ att,
                                              const int* __restrict__ lengths,
                                              float* __restrict__ pooled) {
  __shared__ float red[256];
  __shared__ float wbuf[Tn];
  int b = blockIdx.x, tid = threadIdx.x;
  int len = lengths[b];
  len = len < 0 ? 0 : (len > Tn ? Tn : len);
  const float* attb = att + (size_t)b * Tn;

  float m = -1e30f;
  for (int t = tid; t < len; t += 256) m = fmaxf(m, attb[t]);
  red[tid] = m; __syncthreads();
  for (int s = 128; s > 0; s >>= 1) {
    if (tid < s) red[tid] = fmaxf(red[tid], red[tid + s]);
    __syncthreads();
  }
  m = red[0]; __syncthreads();

  float ssum = 0.0f;
  for (int t = tid; t < Tn; t += 256) {
    float w = (t < len) ? expf(attb[t] - m) : 0.0f;
    wbuf[t] = w;
    ssum += w;
  }
  red[tid] = ssum; __syncthreads();
  for (int s = 128; s > 0; s >>= 1) {
    if (tid < s) red[tid] += red[tid + s];
    __syncthreads();
  }
  float inv = 1.0f / red[0]; __syncthreads();

  int h = tid & 63, q = tid >> 6;
  float acc = 0.0f;
  for (int t = q; t < Tn; t += 4)
    acc += wbuf[t] * fused[((size_t)b * Tn + t) * 64 + h];
  red[tid] = acc; __syncthreads();
  if (tid < 64)
    pooled[b * 64 + tid] =
        (red[tid] + red[64 + tid] + red[128 + tid] + red[192 + tid]) * inv;
}

// tiny classifier head: 64 -> 64 -> 32 -> 1 per batch row
__global__ void __launch_bounds__(512) k_cls(const float* __restrict__ pooled,
                                             const float* __restrict__ w1,
                                             const float* __restrict__ b1,
                                             const float* __restrict__ w2,
                                             const float* __restrict__ b2,
                                             const float* __restrict__ w3,
                                             const float* __restrict__ b3,
                                             float* __restrict__ out) {
  __shared__ float z1[8][64];
  __shared__ float z2[8][32];
  int tid = threadIdx.x, b = tid >> 6, o = tid & 63;
  float a = b1[o];
  for (int c = 0; c < 64; ++c) a += w1[o * 64 + c] * pooled[b * 64 + c];
  z1[b][o] = fmaxf(a, 0.0f);
  __syncthreads();
  if (o < 32) {
    float a2 = b2[o];
    for (int c = 0; c < 64; ++c) a2 += w2[o * 64 + c] * z1[b][c];
    z2[b][o] = fmaxf(a2, 0.0f);
  }
  __syncthreads();
  if (o == 0) {
    float a3 = b3[0];
    for (int c = 0; c < 32; ++c) a3 += w3[c] * z2[b][c];
    out[b] = a3;
  }
}

// ===========================================================================
extern "C" void kernel_launch(void* const* d_in, const int* in_sizes, int n_in,
                              void* d_out, int out_size, void* d_ws, size_t ws_size,
                              hipStream_t stream) {
  (void)in_sizes; (void)n_in; (void)out_size; (void)ws_size;
  auto F = [&](int i) { return (const float*)d_in[i]; };

  const float* x = F(0);
  const int* lengths = (const int*)d_in[1];
  const float* in_proj_w = F(2);
  const float* in_proj_b = F(3);
  int p = 4 + 30 * 8;
  const float* post_w1 = F(p + 0); const float* post_b1 = F(p + 1);
  const float* post_w2 = F(p + 2); const float* post_b2 = F(p + 3);
  const float* Wih0 = F(p + 4); const float* Whh0 = F(p + 5);
  const float* bih0 = F(p + 6); const float* bhh0 = F(p + 7);
  const float* Wih1 = F(p + 8); const float* Whh1 = F(p + 9);
  const float* bih1 = F(p + 10); const float* bhh1 = F(p + 11);
  const float* fus_w1 = F(p + 12); const float* fus_b1 = F(p + 13);
  const float* fus_w2 = F(p + 14); const float* fus_b2 = F(p + 15);
  const float* att_w1 = F(p + 16); const float* att_b1 = F(p + 17);
  const float* att_w2 = F(p + 18); const float* att_b2 = F(p + 19);
  const float* cls_w1 = F(p + 20); const float* cls_b1 = F(p + 21);
  const float* cls_w2 = F(p + 22); const float* cls_b2 = F(p + 23);
  const float* cls_w3 = F(p + 24); const float* cls_b3 = F(p + 25);

  // ---- workspace layout (units of U = B*H*T = 2M floats = 8 MB) ----------
  float* ws = (float*)d_ws;
  const size_t U = (size_t)Bn * Hn * Tn;    // 2,097,152
  float* bufA = ws + 0 * U;                 // resid ping (later: fused f32)
  float* bufB = ws + 1 * U;                 // resid pong (later: att + pooled)
  float* skip = ws + 2 * U;
  _Float16* hwav_h = (_Float16*)(ws + 3 * U);          // U f16, A-packed
  _Float16* l1_h   = hwav_h + U;                       // U f16, A-packed
  float* xg = ws + 4 * U;                   // 4 units: [T][8][256] f32
  _Float16* l2_h    = (_Float16*)(ws + 8 * U);         // U f16, A-packed
  _Float16* fused_h = l2_h + U;                        // U f16, B-packed
  _Float16* mA = (_Float16*)(ws + 9 * U);              // resid mirrors, B-packed
  _Float16* mB = mA + U;
  _Float16* pk = (_Float16*)(ws + 10 * U);  // packed weights (~1.7 MB)
  float* fused  = bufA;
  float* attbuf = bufB;
  float* pooled = bufB + 65536;

  size_t po = 0;
  auto apk = [&](size_t n) { _Float16* q = pk + po; po += n; return q; };
  _Float16* pkBlk = apk((size_t)30 * 6 * 4096);
  _Float16* pkIn  = apk(2048);
  _Float16* pkP1  = apk(4096);
  _Float16* pkP2  = apk(4096);
  _Float16* pkA1  = apk(4096);
  _Float16* pkWih0 = apk(16384); _Float16* pkWhh0 = apk(16384);
  _Float16* pkWih1 = apk(16384); _Float16* pkWhh1 = apk(16384);
  _Float16* pkF1  = apk(12288);
  _Float16* pkF2  = apk(4096);

  // ---- one-time weight pre-swizzle into WMMA lane layout ------------------
  k_packA<<<1, 256, 0, stream>>>(in_proj_w, 32, 1, 4, 1, pkIn);
  for (int i = 0; i < 30; ++i) {
    const float* fw = F(4 + 8 * i + 0);
    const float* gw = F(4 + 8 * i + 2);
    const float* rw = F(4 + 8 * i + 4);
    const float* sw = F(4 + 8 * i + 6);
    _Float16* base = pkBlk + (size_t)i * 6 * 4096;
    k_packA<<<1, 256, 0, stream>>>(fw + 1, 128, 2, 4, 2, base + 0 * 4096);
    k_packA<<<1, 256, 0, stream>>>(fw + 0, 128, 2, 4, 2, base + 1 * 4096);
    k_packA<<<1, 256, 0, stream>>>(gw + 1, 128, 2, 4, 2, base + 2 * 4096);
    k_packA<<<1, 256, 0, stream>>>(gw + 0, 128, 2, 4, 2, base + 3 * 4096);
    k_packA<<<1, 256, 0, stream>>>(rw, 64, 1, 4, 2, base + 4 * 4096);
    k_packA<<<1, 256, 0, stream>>>(sw, 64, 1, 4, 2, base + 5 * 4096);
  }
  k_packA<<<1, 256, 0, stream>>>(post_w1, 64, 1, 4, 2, pkP1);
  k_packA<<<1, 256, 0, stream>>>(post_w2, 64, 1, 4, 2, pkP2);
  k_packA<<<1, 256, 0, stream>>>(att_w1, 64, 1, 4, 2, pkA1);
  k_packB<<<4, 256, 0, stream>>>(Wih0, 1, 64, 2, 16, pkWih0);
  k_packB<<<4, 256, 0, stream>>>(Whh0, 1, 64, 2, 16, pkWhh0);
  k_packB<<<4, 256, 0, stream>>>(Wih1, 1, 64, 2, 16, pkWih1);
  k_packB<<<4, 256, 0, stream>>>(Whh1, 1, 64, 2, 16, pkWhh1);
  k_packB<<<3, 256, 0, stream>>>(fus_w1, 1, 192, 6, 4, pkF1);
  k_packB<<<1, 256, 0, stream>>>(fus_w2, 1, 64, 2, 4, pkF2);

  // ---- pipeline -----------------------------------------------------------
  const int gridBT = Bn * (Tn / 64);        // 512
  k_zero<<<(int)((U + 255) / 256), 256, 0, stream>>>(skip, (int)U);
  k_inproj<<<gridBT, 256, 0, stream>>>(x, pkIn, in_proj_b, bufA, mA);

  for (int i = 0; i < 30; ++i) {
    const float* fb = F(4 + 8 * i + 1);
    const float* gb = F(4 + 8 * i + 3);
    const float* rb = F(4 + 8 * i + 5);
    const float* sb = F(4 + 8 * i + 7);
    float* rin  = (i & 1) ? bufB : bufA;
    float* rout = (i & 1) ? bufA : bufB;
    _Float16* rin_h  = (i & 1) ? mB : mA;
    _Float16* rout_h = (i & 1) ? mA : mB;
    int dil = 1 << (i % 10);
    k_block<<<gridBT, 256, 0, stream>>>(rin, rin_h, rout, rout_h, skip,
                                        pkBlk + (size_t)i * 6 * 4096,
                                        fb, gb, rb, sb, dil);
  }

  k_post<<<gridBT, 256, 0, stream>>>(skip, pkP1, post_b1, pkP2, post_b2, hwav_h);

  k_xg<<<Pn / 32, 256, 0, stream>>>(hwav_h, pkWih0, bih0, bhh0, xg);
  k_lstm_rec<<<1, 512, 0, stream>>>(xg, pkWhh0, l1_h);
  k_xg<<<Pn / 32, 256, 0, stream>>>(l1_h, pkWih1, bih1, bhh1, xg);
  k_lstm_rec<<<1, 512, 0, stream>>>(xg, pkWhh1, l2_h);

  k_fusion<<<Pn / 64, 256, 0, stream>>>(hwav_h, l1_h, l2_h, pkF1, fus_b1,
                                        pkF2, fus_b2, fused, fused_h);
  k_att<<<Pn / 64, 256, 0, stream>>>(fused_h, pkA1, att_b1, att_w2, att_b2, attbuf);
  k_pool<<<Bn, 256, 0, stream>>>(fused, attbuf, lengths, pooled);
  k_cls<<<1, 512, 0, stream>>>(pooled, cls_w1, cls_b1, cls_w2, cls_b2,
                               cls_w3, cls_b3, (float*)d_out);
}
